// SolverHead_81140522156594
// MI455X (gfx1250) — matrix operand order repaired and checked by reading, not compiled, not run on hardware
//
#include <hip/hip_runtime.h>
#include <hip/hip_bf16.h>
#include <math.h>

// ---------------- problem constants (match reference) ----------------
#define DMODEL 512
#define NHEAD  8
#define DHEAD  64
#define NLAYER 2
#define BATCH  64
#define SEQ    256
#define CTXLEN 128
#define NSLOT  8
#define VOCAB  512

// ---------------- types ----------------
typedef __bf16 bf16_t;
typedef __bf16 v16bf __attribute__((ext_vector_type(16)));
typedef __bf16 v8bf  __attribute__((ext_vector_type(8)));
typedef float  v8f   __attribute__((ext_vector_type(8)));
typedef float  v4f   __attribute__((ext_vector_type(4)));

__device__ __forceinline__ bf16_t f2bf(float x) {
  union { float f; unsigned u; } a; a.f = x;
  unsigned r = a.u + 0x7FFFu + ((a.u >> 16) & 1u);   // round-to-nearest-even
  unsigned short hs = (unsigned short)(r >> 16);
  bf16_t b; __builtin_memcpy(&b, &hs, 2);
  return b;
}

__device__ __forceinline__ float softplusf(float x) {
  return (x > 20.0f) ? x : log1pf(expf(x));
}

// ---------------- generic bf16 WMMA GEMM ----------------
// C[m,n] = alpha * sum_k A[m,k]*W[n,k] + bias[n]      (bias always valid)
// A element (m,k) at Ab + m*lda + k*aStrideK
// W element (n,k) at Wb + n*ldwN + k*wStrideK   (W == B^T of the math GEMM)
// per-z-batch bases: X + (bz/Hdiv)*xOffB + (bz%Hdiv)*xOffH
#define BM  128
#define BN  128
#define BKK 32
#define LDT 40   // padded LDS row stride (elements); 80B, 16B-aligned chunks

__global__ __launch_bounds__(256)
void gemm_bf16_wmma(const float* __restrict__ A, const float* __restrict__ W,
                    const float* __restrict__ bias, float* __restrict__ C,
                    int M, int N, int K,
                    long long aOffB, long long aOffH, int lda, int aStrideK,
                    long long wOffB, long long wOffH, int ldwN, int wStrideK,
                    long long cOffB, long long cOffH, int ldc,
                    int Hdiv, float alpha)
{
  __shared__ __align__(16) bf16_t As[BM * LDT];
  __shared__ __align__(16) bf16_t Ws[BN * LDT];

  const int tid  = threadIdx.x;
  const int lane = tid & 31;
  const int wave = tid >> 5;
  const int half = lane >> 4;
  const int ln15 = lane & 15;

  const int bz = blockIdx.z;
  const int bb = bz / Hdiv;
  const int hh = bz - bb * Hdiv;
  const float* Ab = A + (long long)bb * aOffB + (long long)hh * aOffH;
  const float* Wb = W + (long long)bb * wOffB + (long long)hh * wOffH;
  float*       Cb = C + (long long)bb * cOffB + (long long)hh * cOffH;

  const int m0 = blockIdx.y * BM;
  const int n0 = blockIdx.x * BN;

  // uniform per-block decisions
  const bool fullT = (m0 + BM <= M) && (n0 + BN <= N);
  const bool fast  = (aStrideK == 1) && (wStrideK == 1) && fullT;

  // 8 waves tiled 4(M) x 2(N): each wave owns 32x64 of the 128x128 tile
  const int wm = (wave >> 1) * 32;
  const int wn = (wave & 1) * 64;

  v8f acc[2][4];
#pragma unroll
  for (int i = 0; i < 2; i++)
#pragma unroll
    for (int j = 0; j < 4; j++)
      acc[i][j] = (v8f){0.f, 0.f, 0.f, 0.f, 0.f, 0.f, 0.f, 0.f};

  // fast-path thread mapping: row = tid/2 (0..127), 16-wide K segment = (tid&1)*16
  const int fr  = tid >> 1;
  const int fc0 = (tid & 1) << 4;

  for (int kt = 0; kt < K; kt += BKK) {
    if (fast) {
      // ---- vectorized fill: 4x float4 per tile per thread, b128 LDS stores ----
      const float* ga = Ab + (long long)(m0 + fr) * lda + (kt + fc0);
      const float* gw = Wb + (long long)(n0 + fr) * ldwN + (kt + fc0);
      v4f a0 = *(const v4f*)(ga + 0);
      v4f a1 = *(const v4f*)(ga + 4);
      v4f a2 = *(const v4f*)(ga + 8);
      v4f a3 = *(const v4f*)(ga + 12);
      v4f w0 = *(const v4f*)(gw + 0);
      v4f w1 = *(const v4f*)(gw + 4);
      v4f w2 = *(const v4f*)(gw + 8);
      v4f w3 = *(const v4f*)(gw + 12);
      if (kt + BKK < K) {
        __builtin_prefetch(ga + BKK, 0, 1);
        __builtin_prefetch(gw + BKK, 0, 1);
      }
      bf16_t ta[16], tw[16];
#pragma unroll
      for (int q = 0; q < 4; q++) {
        ta[q]      = f2bf(a0[q]); ta[4 + q]  = f2bf(a1[q]);
        ta[8 + q]  = f2bf(a2[q]); ta[12 + q] = f2bf(a3[q]);
        tw[q]      = f2bf(w0[q]); tw[4 + q]  = f2bf(w1[q]);
        tw[8 + q]  = f2bf(w2[q]); tw[12 + q] = f2bf(w3[q]);
      }
      *(v8bf*)(&As[fr * LDT + fc0])     = *(v8bf*)(&ta[0]);
      *(v8bf*)(&As[fr * LDT + fc0 + 8]) = *(v8bf*)(&ta[8]);
      *(v8bf*)(&Ws[fr * LDT + fc0])     = *(v8bf*)(&tw[0]);
      *(v8bf*)(&Ws[fr * LDT + fc0 + 8]) = *(v8bf*)(&tw[8]);
    } else {
      // ---- branch-free scalar fill (edge tiles / strided-K operands) ----
      // clamp row to a legal address, select 0 afterwards; single rolled copy
#pragma unroll 1
      for (int t = 0; t < 16; t++) {
        int e = tid + t * 256;
        int r = e >> 5;          // 0..127
        int c = e & 31;          // 0..31
        int gm = m0 + r;
        int gmc = (gm < M) ? gm : (M - 1);
        float av = Ab[(long long)gmc * lda + (long long)(kt + c) * aStrideK];
        av = (gm < M) ? av : 0.0f;
        As[r * LDT + c] = f2bf(av);
        int gn = n0 + r;
        int gnc = (gn < N) ? gn : (N - 1);
        float wv = Wb[(long long)gnc * ldwN + (long long)(kt + c) * wStrideK];
        wv = (gn < N) ? wv : 0.0f;
        Ws[r * LDT + c] = f2bf(wv);
      }
    }
    __syncthreads();

    // fragment loads per CDNA5 WMMA VGPR layouts
    union Fr { v16bf v; v8bf h[2]; };
    Fr fa[2], fb[4];
#pragma unroll
    for (int i = 0; i < 2; i++) {
      const bf16_t* p = &As[(wm + i * 16 + ln15) * LDT];
      fa[i].h[0] = *(const v8bf*)(p + half * 8);        // K 0..7  / 8..15
      fa[i].h[1] = *(const v8bf*)(p + 16 + half * 8);   // K 16..23 / 24..31
    }
#pragma unroll
    for (int j = 0; j < 4; j++) {
      const bf16_t* p = &Ws[(wn + j * 16 + ln15) * LDT];
      fb[j].h[0] = *(const v8bf*)(p + half * 8);        // K 0..7  / 8..15
      fb[j].h[1] = *(const v8bf*)(p + 16 + half * 8);   // K 16..23 / 24..31
    }
#pragma unroll
    for (int i = 0; i < 2; i++)
#pragma unroll
      for (int j = 0; j < 4; j++)
        acc[i][j] = __builtin_amdgcn_wmma_f32_16x16x32_bf16(
            false, fa[i].v, false, fb[j].v, (short)0, acc[i][j], false, false);
    __syncthreads();
  }

  // epilogue: C/D layout — VGPR r: M = r + (half?8:0), N = lane&15
  if (fullT) {
#pragma unroll
    for (int i = 0; i < 2; i++) {
#pragma unroll
      for (int j = 0; j < 4; j++) {
        int gn = n0 + wn + j * 16 + ln15;
        float bv = bias[gn];
#pragma unroll
        for (int r = 0; r < 8; r++) {
          int gm = m0 + wm + i * 16 + half * 8 + r;
          Cb[(long long)gm * ldc + gn] = acc[i][j][r] * alpha + bv;
        }
      }
    }
  } else {
#pragma unroll 1
    for (int i = 0; i < 2; i++) {
#pragma unroll 1
      for (int j = 0; j < 4; j++) {
        int gn = n0 + wn + j * 16 + ln15;
        if (gn >= N) continue;
        float bv = bias[gn];
#pragma unroll 1
        for (int r = 0; r < 8; r++) {
          int gm = m0 + wm + i * 16 + half * 8 + r;
          if (gm >= M) continue;
          Cb[(long long)gm * ldc + gn] = acc[i][j][r] * alpha + bv;
        }
      }
    }
  }
}

// ---------------- zero-fill small vector (for no-bias GEMMs) ----------------
__global__ void zero_kernel(float* __restrict__ p, int n)
{
  int i = blockIdx.x * blockDim.x + threadIdx.x;
  if (i < n) p[i] = 0.0f;
}

// ---------------- elementwise activation: 1=relu, 2=gelu(exact) ----------------
__global__ __launch_bounds__(256)
void act_kernel(float* __restrict__ x, long long n, int mode)
{
  long long i0 = ((long long)blockIdx.x * 256 + threadIdx.x) * 4;
  if (i0 + 3 >= n) {
    for (long long i = i0; i < n; i++) {
      float v = x[i];
      x[i] = (mode == 1) ? fmaxf(v, 0.0f)
                         : 0.5f * v * (1.0f + erff(v * 0.7071067811865476f));
    }
    return;
  }
  v4f v = *(v4f*)(x + i0);
  if (mode == 1) {
#pragma unroll
    for (int q = 0; q < 4; q++) v[q] = fmaxf(v[q], 0.0f);
  } else {
#pragma unroll
    for (int q = 0; q < 4; q++) v[q] = 0.5f * v[q] * (1.0f + erff(v[q] * 0.7071067811865476f));
  }
  *(v4f*)(x + i0) = v;
}

// ---------------- slot softmax + fuse-input concat ----------------
__global__ void slot_fuse_kernel(const float* __restrict__ ctx_sum,
                                 const float* __restrict__ a_slots,
                                 const float* __restrict__ a_scores,
                                 const float* __restrict__ r_slots,
                                 const float* __restrict__ r_scores,
                                 float* __restrict__ fuse_in)
{
  int b = blockIdx.x;
  float wa[NSLOT], wr[NSLOT];
  float ma = -1e30f, mr = -1e30f;
  for (int s = 0; s < NSLOT; s++) {
    ma = fmaxf(ma, a_scores[b * NSLOT + s]);
    mr = fmaxf(mr, r_scores[b * NSLOT + s]);
  }
  float sa = 0.f, sr = 0.f;
  for (int s = 0; s < NSLOT; s++) {
    wa[s] = expf(a_scores[b * NSLOT + s] - ma); sa += wa[s];
    wr[s] = expf(r_scores[b * NSLOT + s] - mr); sr += wr[s];
  }
  for (int s = 0; s < NSLOT; s++) { wa[s] /= sa; wr[s] /= sr; }

  float* o = fuse_in + (long long)b * 3 * DMODEL;
  for (int d = threadIdx.x; d < DMODEL; d += blockDim.x) {
    float as = 0.f, rs = 0.f;
    for (int s = 0; s < NSLOT; s++) {
      as += a_slots[((long long)b * NSLOT + s) * DMODEL + d] * wa[s];
      rs += r_slots[((long long)b * NSLOT + s) * DMODEL + d] * wr[s];
    }
    o[d]              = ctx_sum[(long long)b * DMODEL + d];
    o[DMODEL + d]     = as;
    o[2 * DMODEL + d] = rs;
  }
}

// ---------------- embedding + positional + cond ----------------
__global__ void embed_kernel(const float* __restrict__ qe, const float* __restrict__ pe,
                             const float* __restrict__ cond, const int* __restrict__ fq,
                             float* __restrict__ h)
{
  int row = blockIdx.x;              // b*SEQ + l
  int b = row / SEQ, l = row - b * SEQ;
  int tok = fq[row];
  const float* e = qe + (long long)tok * DMODEL;
  const float* p = pe + (long long)l * DMODEL;
  const float* c = cond + (long long)b * DMODEL;
  float* o = h + (long long)row * DMODEL;
  for (int d = threadIdx.x; d < DMODEL; d += blockDim.x)
    o[d] = e[d] + p[d] + c[d];
}

// ---------------- row softmax (in==out safe) ----------------
__global__ __launch_bounds__(256)
void softmax_rows(const float* __restrict__ in, float* __restrict__ out, int len)
{
  __shared__ float red[256];
  long long row = blockIdx.x;
  const float* x = in + row * len;
  float* y = out + row * len;
  int tid = threadIdx.x;
  float m = -3.4e38f;
  for (int i = tid; i < len; i += 256) m = fmaxf(m, x[i]);
  red[tid] = m; __syncthreads();
  for (int s = 128; s > 0; s >>= 1) { if (tid < s) red[tid] = fmaxf(red[tid], red[tid + s]); __syncthreads(); }
  m = red[0]; __syncthreads();
  float sum = 0.f;
  for (int i = tid; i < len; i += 256) { float e = expf(x[i] - m); y[i] = e; sum += e; }
  red[tid] = sum; __syncthreads();
  for (int s = 128; s > 0; s >>= 1) { if (tid < s) red[tid] += red[tid + s]; __syncthreads(); }
  float inv = 1.0f / red[0];
  for (int i = tid; i < len; i += 256) y[i] *= inv;
}

// ---------------- (a [+ b]) -> LayerNorm(scale,bias) ----------------
__global__ __launch_bounds__(256)
void add_ln_kernel(const float* __restrict__ a, const float* __restrict__ b,
                   const float* __restrict__ s, const float* __restrict__ bi,
                   float* __restrict__ out)
{
  __shared__ float sh[DMODEL];
  __shared__ float red[256];
  long long row = blockIdx.x;
  int tid = threadIdx.x;
  const float* pa = a + row * DMODEL;
  const float* pb = b ? (b + row * DMODEL) : nullptr;
  float lsum = 0.f;
  for (int d = tid; d < DMODEL; d += 256) {
    float x = pa[d] + (pb ? pb[d] : 0.0f);
    sh[d] = x; lsum += x;
  }
  red[tid] = lsum; __syncthreads();
  for (int st = 128; st > 0; st >>= 1) { if (tid < st) red[tid] += red[tid + st]; __syncthreads(); }
  float mean = red[0] * (1.0f / DMODEL); __syncthreads();
  float lvar = 0.f;
  for (int d = tid; d < DMODEL; d += 256) { float dv = sh[d] - mean; lvar += dv * dv; }
  red[tid] = lvar; __syncthreads();
  for (int st = 128; st > 0; st >>= 1) { if (tid < st) red[tid] += red[tid + st]; __syncthreads(); }
  float inv = rsqrtf(red[0] * (1.0f / DMODEL) + 1e-5f);
  for (int d = tid; d < DMODEL; d += 256)
    out[row * DMODEL + d] = (sh[d] - mean) * inv * s[d] + bi[d];
}

// ---------------- final logit combine (adds into d_out) ----------------
__global__ void combine_kernel(const float* __restrict__ prior, const float* __restrict__ sprobs,
                               const int* __restrict__ fq,
                               const float* __restrict__ shift_scale,
                               const float* __restrict__ demo_scale,
                               float* __restrict__ out)
{
  long long row = blockIdx.x;        // b*SEQ + l
  int b = (int)(row / SEQ);
  int q0 = fq[row];
  float sd = softplusf(demo_scale[0]);
  float ss = softplusf(shift_scale[0]);
  const float* pr = prior + (long long)b * VOCAB;
  const float* sp = sprobs + row * VOCAB;
  float* o = out + row * VOCAB;
  for (int v = threadIdx.x; v < VOCAB; v += blockDim.x) {
    int idx = v - q0; if (idx < 0) idx += VOCAB;
    o[v] += sd * logf(fmaxf(pr[idx], 1e-8f)) + ss * logf(fmaxf(sp[idx], 1e-8f));
  }
}

// ---------------- host side ----------------
static inline void launch_gemm(hipStream_t s,
    const float* A, const float* W, const float* bias, float* C,
    int M, int N, int K,
    long long aOffB, long long aOffH, int lda, int aStrideK,
    long long wOffB, long long wOffH, int ldwN, int wStrideK,
    long long cOffB, long long cOffH, int ldc,
    int batch, int Hdiv, float alpha)
{
  dim3 grid((N + BN - 1) / BN, (M + BM - 1) / BM, batch);
  gemm_bf16_wmma<<<grid, 256, 0, s>>>(A, W, bias, C, M, N, K,
      aOffB, aOffH, lda, aStrideK, wOffB, wOffH, ldwN, wStrideK,
      cOffB, cOffH, ldc, Hdiv, alpha);
}

extern "C" void kernel_launch(void* const* d_in, const int* in_sizes, int n_in,
                              void* d_out, int out_size, void* d_ws, size_t ws_size,
                              hipStream_t stream)
{
  (void)in_sizes; (void)n_in; (void)out_size; (void)ws_size;
  const float* query_emb   = (const float*)d_in[0];
  const float* pos_emb     = (const float*)d_in[1];
  const float* fuse_w1     = (const float*)d_in[2];
  const float* fuse_b1     = (const float*)d_in[3];
  const float* fuse_w2     = (const float*)d_in[4];
  const float* fuse_b2     = (const float*)d_in[5];
  const float* enc_in_w    = (const float*)d_in[6];
  const float* enc_in_b    = (const float*)d_in[7];
  const float* enc_out_w   = (const float*)d_in[8];
  const float* enc_out_b   = (const float*)d_in[9];
  const float* enc_l1_w    = (const float*)d_in[10];
  const float* enc_l1_b    = (const float*)d_in[11];
  const float* enc_l2_w    = (const float*)d_in[12];
  const float* enc_l2_b    = (const float*)d_in[13];
  const float* enc_ln1_s   = (const float*)d_in[14];
  const float* enc_ln1_b   = (const float*)d_in[15];
  const float* enc_ln2_s   = (const float*)d_in[16];
  const float* enc_ln2_b   = (const float*)d_in[17];
  const float* ctx_in_w    = (const float*)d_in[18];
  const float* ctx_in_b    = (const float*)d_in[19];
  const float* ctx_out_w   = (const float*)d_in[20];
  const float* ctx_out_b   = (const float*)d_in[21];
  const float* norm_s      = (const float*)d_in[22];
  const float* norm_b      = (const float*)d_in[23];
  const float* out_w       = (const float*)d_in[24];
  const float* out_b       = (const float*)d_in[25];
  const float* shift_w     = (const float*)d_in[26];
  const float* shift_b     = (const float*)d_in[27];
  const float* shift_scale = (const float*)d_in[28];
  const float* demo_scale  = (const float*)d_in[29];
  const float* ctx_summary = (const float*)d_in[30];
  const float* a_slots     = (const float*)d_in[31];
  const float* a_scores    = (const float*)d_in[32];
  const float* r_slots     = (const float*)d_in[33];
  const float* r_scores    = (const float*)d_in[34];
  const float* ctx_tokens  = (const float*)d_in[35];
  const float* demo_logp   = (const float*)d_in[36];
  const int*   final_query = (const int*)d_in[37];
  float* out = (float*)d_out;

  // workspace layout (floats)
  float* ws = (float*)d_ws;
  size_t off = 0;
  float* zbias    = ws + off; off += 2048;                                // zero "bias" for no-bias GEMMs
  float* fuse_in  = ws + off; off += (size_t)BATCH * 3 * DMODEL;          // 98304
  float* fuse_hid = ws + off; off += (size_t)BATCH * 2 * DMODEL;          // 65536
  float* cond     = ws + off; off += (size_t)BATCH * DMODEL;              // 32768
  float* h        = ws + off; off += (size_t)BATCH * SEQ * DMODEL;        // 8.4M
  float* big1     = ws + off; off += (size_t)BATCH * SEQ * 4 * DMODEL;    // 33.5M (qkv / ffn-hidden / tmp2)
  float* scores   = ws + off; off += (size_t)BATCH * NHEAD * SEQ * SEQ;   // 33.5M (also shift logits)
  float* tmp      = ws + off; off += (size_t)BATCH * SEQ * DMODEL;        // 8.4M (attn-out / ff-out)
  float* ctx_kv   = ws + off; off += (size_t)BATCH * CTXLEN * 2 * DMODEL; // 8.4M
  float* prior    = ws + off; off += (size_t)BATCH * VOCAB;               // 32768

  const int ROWS = BATCH * SEQ;       // 16384
  const int BH   = BATCH * NHEAD;     // 512
  const float iscale = 0.125f;        // 1/sqrt(64)

  // 0) zero bias vector
  zero_kernel<<<8, 256, 0, stream>>>(zbias, 2048);
  // 1) slot softmax + fuse input
  slot_fuse_kernel<<<BATCH, 256, 0, stream>>>(ctx_summary, a_slots, a_scores, r_slots, r_scores, fuse_in);
  // 2) cond = gelu(fuse_in @ w1^T + b1) @ w2^T + b2
  launch_gemm(stream, fuse_in, fuse_w1, fuse_b1, fuse_hid, BATCH, 2*DMODEL, 3*DMODEL,
              0,0, 3*DMODEL,1, 0,0, 3*DMODEL,1, 0,0, 2*DMODEL, 1,1, 1.0f);
  act_kernel<<<(BATCH*2*DMODEL/4 + 255)/256, 256, 0, stream>>>(fuse_hid, (long long)BATCH*2*DMODEL, 2);
  launch_gemm(stream, fuse_hid, fuse_w2, fuse_b2, cond, BATCH, DMODEL, 2*DMODEL,
              0,0, 2*DMODEL,1, 0,0, 2*DMODEL,1, 0,0, DMODEL, 1,1, 1.0f);
  // 3) h = emb[fq] + pos + cond
  embed_kernel<<<ROWS, 256, 0, stream>>>(query_emb, pos_emb, cond, final_query, h);

  // 4) encoder layers
  for (int i = 0; i < NLAYER; i++) {
    const float* wIn  = enc_in_w  + (size_t)i * 3 * DMODEL * DMODEL;
    const float* bIn  = enc_in_b  + (size_t)i * 3 * DMODEL;
    const float* wOut = enc_out_w + (size_t)i * DMODEL * DMODEL;
    const float* bOut = enc_out_b + (size_t)i * DMODEL;
    const float* w1   = enc_l1_w  + (size_t)i * 4 * DMODEL * DMODEL;
    const float* b1   = enc_l1_b  + (size_t)i * 4 * DMODEL;
    const float* w2   = enc_l2_w  + (size_t)i * DMODEL * 4 * DMODEL;
    const float* b2   = enc_l2_b  + (size_t)i * DMODEL;

    // qkv = h @ wIn^T + bIn   -> big1 [B,L,3D]
    launch_gemm(stream, h, wIn, bIn, big1, ROWS, 3*DMODEL, DMODEL,
                0,0, DMODEL,1, 0,0, DMODEL,1, 0,0, 3*DMODEL, 1,1, 1.0f);
    // scores[b,h] = (q @ k^T)/8 -> scores [B,H,L,L]
    launch_gemm(stream, big1, big1 + DMODEL, zbias, scores, SEQ, SEQ, DHEAD,
                (long long)SEQ*3*DMODEL, DHEAD, 3*DMODEL, 1,
                (long long)SEQ*3*DMODEL, DHEAD, 3*DMODEL, 1,
                (long long)NHEAD*SEQ*SEQ, (long long)SEQ*SEQ, SEQ,
                BH, NHEAD, iscale);
    softmax_rows<<<BH*SEQ, 256, 0, stream>>>(scores, scores, SEQ);
    // o[b,h] = P @ v  -> tmp [B,L,D] (heads merged)
    launch_gemm(stream, scores, big1 + 2*DMODEL, zbias, tmp, SEQ, DHEAD, SEQ,
                (long long)NHEAD*SEQ*SEQ, (long long)SEQ*SEQ, SEQ, 1,
                (long long)SEQ*3*DMODEL, DHEAD, 1, 3*DMODEL,
                (long long)SEQ*DMODEL, DHEAD, DMODEL,
                BH, NHEAD, 1.0f);
    // sa = o @ wOut^T + bOut -> big1
    launch_gemm(stream, tmp, wOut, bOut, big1, ROWS, DMODEL, DMODEL,
                0,0, DMODEL,1, 0,0, DMODEL,1, 0,0, DMODEL, 1,1, 1.0f);
    add_ln_kernel<<<ROWS, 256, 0, stream>>>(h, big1, enc_ln1_s + (size_t)i*DMODEL, enc_ln1_b + (size_t)i*DMODEL, h);
    // ffn
    launch_gemm(stream, h, w1, b1, big1, ROWS, 4*DMODEL, DMODEL,
                0,0, DMODEL,1, 0,0, DMODEL,1, 0,0, 4*DMODEL, 1,1, 1.0f);
    act_kernel<<<((long long)ROWS*4*DMODEL/4 + 255)/256, 256, 0, stream>>>(big1, (long long)ROWS*4*DMODEL, 1);
    launch_gemm(stream, big1, w2, b2, tmp, ROWS, DMODEL, 4*DMODEL,
                0,0, 4*DMODEL,1, 0,0, 4*DMODEL,1, 0,0, DMODEL, 1,1, 1.0f);
    add_ln_kernel<<<ROWS, 256, 0, stream>>>(h, tmp, enc_ln2_s + (size_t)i*DMODEL, enc_ln2_b + (size_t)i*DMODEL, h);
  }

  // 5) cross attention with context tokens
  // q = h @ wq^T + bq -> big1 [B,L,D]
  launch_gemm(stream, h, ctx_in_w, ctx_in_b, big1, ROWS, DMODEL, DMODEL,
              0,0, DMODEL,1, 0,0, DMODEL,1, 0,0, DMODEL, 1,1, 1.0f);
  // kv = ctx @ w[kv]^T + b[kv] -> ctx_kv [B,CTX,2D]
  launch_gemm(stream, ctx_tokens, ctx_in_w + (size_t)DMODEL*DMODEL, ctx_in_b + DMODEL, ctx_kv,
              BATCH*CTXLEN, 2*DMODEL, DMODEL,
              0,0, DMODEL,1, 0,0, DMODEL,1, 0,0, 2*DMODEL, 1,1, 1.0f);
  // scores[b,h] = (q @ k^T)/8 -> scores [B,H,L,CTX]
  launch_gemm(stream, big1, ctx_kv, zbias, scores, SEQ, CTXLEN, DHEAD,
              (long long)SEQ*DMODEL, DHEAD, DMODEL, 1,
              (long long)CTXLEN*2*DMODEL, DHEAD, 2*DMODEL, 1,
              (long long)NHEAD*SEQ*CTXLEN, (long long)SEQ*CTXLEN, CTXLEN,
              BH, NHEAD, iscale);
  softmax_rows<<<BH*SEQ, 256, 0, stream>>>(scores, scores, CTXLEN);
  // o = P @ v -> tmp
  launch_gemm(stream, scores, ctx_kv + DMODEL, zbias, tmp, SEQ, DHEAD, CTXLEN,
              (long long)NHEAD*SEQ*CTXLEN, (long long)SEQ*CTXLEN, CTXLEN, 1,
              (long long)CTXLEN*2*DMODEL, DHEAD, 1, 2*DMODEL,
              (long long)SEQ*DMODEL, DHEAD, DMODEL,
              BH, NHEAD, 1.0f);
  // sa = o @ ctx_out_w^T + b -> big1
  launch_gemm(stream, tmp, ctx_out_w, ctx_out_b, big1, ROWS, DMODEL, DMODEL,
              0,0, DMODEL,1, 0,0, DMODEL,1, 0,0, DMODEL, 1,1, 1.0f);
  add_ln_kernel<<<ROWS, 256, 0, stream>>>(h, big1, norm_s, norm_b, h);

  // 6) head: base logits -> d_out; shift logits -> scores buffer
  launch_gemm(stream, h, out_w, out_b, out, ROWS, VOCAB, DMODEL,
              0,0, DMODEL,1, 0,0, DMODEL,1, 0,0, VOCAB, 1,1, 1.0f);
  launch_gemm(stream, h, shift_w, shift_b, scores, ROWS, VOCAB, DMODEL,
              0,0, DMODEL,1, 0,0, DMODEL,1, 0,0, VOCAB, 1,1, 1.0f);
  softmax_rows<<<BATCH, 256, 0, stream>>>(demo_logp, prior, VOCAB);
  softmax_rows<<<ROWS, 256, 0, stream>>>(scores, scores, VOCAB);
  combine_kernel<<<ROWS, 256, 0, stream>>>(prior, scores, final_query, shift_scale, demo_scale, out);
}
